// CrossAttentionPooler_59090160059100
// MI455X (gfx1250) — compile-verified, matrix operand order checked by reading, not run on hardware
//
#include <hip/hip_runtime.h>
#include <math.h>

// ---------------------------------------------------------------------------
// Bahdanau additive cross-attention pooling, fused for MI455X (gfx1250).
//   N=32, TQ=16, TK=2048, D=128.
// Never materializes h=[N,TQ,TK,D] (536MB). Softmax axis is TQ -> local per
// key column -> full fusion of score/softmax/weighted-sum over k.
// - GEMMs on v_wmma_f32_16x16x32_f16 (f32 accumulate)
// - k tiles staged via GLOBAL_LOAD_ASYNC_TO_LDS_B128 (ASYNCcnt) into an LDS
//   buffer that is reused for uk one phase later (zero extra LDS)
// - tanh on the TRANS pipe via v_tanh_f32 (dominant cost: ~134M tanh)
// ---------------------------------------------------------------------------

#define DD   128
#define NB   32
#define TQ   16
#define TK   2048
#define KCH  8          // k-chunks per n
#define CHUNK (TK/KCH)  // 256 keys per block
#define IC   32         // keys per inner tile (one WMMA K=32 block)

typedef _Float16 v16h __attribute__((ext_vector_type(16)));
typedef _Float16 v8h  __attribute__((ext_vector_type(8)));
typedef _Float16 v4h  __attribute__((ext_vector_type(4)));
typedef float    v8f  __attribute__((ext_vector_type(8)));
typedef float    v4f  __attribute__((ext_vector_type(4)));

// LDS row strides chosen so every 16B fragment run is 16B-aligned and
// lane-strided reads hit distinct banks (row bytes are odd multiples of 16).
#define KL_STR  136   // f16, k chunk row-major   (272B/row)
#define KT_STR  40    // f16, k chunk transposed  (80B/row)
#define WQ_STR  132   // f32, wq                  (528B/row)
#define UK_STR  132   // f32, raw-k stage / uk    (528B/row)
#define WA_STR  40    // f16, softmax weights     (80B/row)

// Native TRANS-pipe tanh (V_TANH_F32, CDNA5). v_nop satisfies the documented
// TRANS hazard (1 independent op before the result may be consumed).
__device__ __forceinline__ float fast_tanh(float x) {
  float y;
  asm("v_tanh_f32 %0, %1\n\tv_nop" : "=v"(y) : "v"(x));
  return y;
}

// Async DMA: global -> LDS, 16B per lane, tracked by ASYNCcnt.
__device__ __forceinline__ void async_copy_b128(unsigned lds_byte_off,
                                                const float* gaddr) {
  asm volatile("global_load_async_to_lds_b128 %0, %1, off"
               :: "v"(lds_byte_off), "v"(gaddr) : "memory");
}
__device__ __forceinline__ void wait_asynccnt0() {
  asm volatile("s_wait_asynccnt 0" ::: "memory");
}

__device__ __forceinline__ v8f wmma_f16(v16h a, v16h b, v8f c) {
  // v_wmma_f32_16x16x32_f16  (probe-verified builtin signature)
  return __builtin_amdgcn_wmma_f32_16x16x32_f16(
      /*neg_a=*/false, a, /*neg_b=*/false, b,
      /*c_mod=*/(short)0, c, /*reuse_a=*/false, /*reuse_b=*/false);
}

// 16-bit A/B fragment lane layout: element K = (v>=4?16:0) + 8*h + 2*(v&3)+s
// => two contiguous runs of 8 halfs at [h*8] and [16+h*8] of the lane's row.
__device__ __forceinline__ v16h frag_h16(const _Float16* row, int h) {
  union { v16h v; v8h p[2]; } u;
  u.p[0] = *(const v8h*)(row + h * 8);
  u.p[1] = *(const v8h*)(row + 16 + h * 8);
  return u.v;
}

__device__ __forceinline__ v16h frag_f32(const float* row, int h) {
  const v4f* r0 = (const v4f*)(row + h * 8);
  const v4f* r1 = (const v4f*)(row + 16 + h * 8);
  v4f a0 = r0[0], a1 = r0[1], b0 = r1[0], b1 = r1[1];
  union { v16h v; _Float16 e[16]; } u;
#pragma unroll
  for (int i = 0; i < 4; ++i) {
    u.e[i]      = (_Float16)a0[i];
    u.e[4 + i]  = (_Float16)a1[i];
    u.e[8 + i]  = (_Float16)b0[i];
    u.e[12 + i] = (_Float16)b1[i];
  }
  return u.v;
}

// --------------------------- kernel 1: Uw -> f16 ---------------------------
__global__ __launch_bounds__(256) void cvt_uw_kernel(const float* __restrict__ Uw,
                                                     _Float16* __restrict__ uwh) {
  int i = blockIdx.x * 256 + threadIdx.x;
  if (i < DD * DD) uwh[i] = (_Float16)Uw[i];
}

// ------------------------ kernel 2: wq = q@Ww.T + Wb -----------------------
// grid = NB blocks, 8 waves; wave w owns the 16-wide d-tile w of wq[n].
__global__ __launch_bounds__(256) void wq_kernel(const float* __restrict__ q,
                                                 const float* __restrict__ Ww,
                                                 const float* __restrict__ Wb,
                                                 float* __restrict__ wq) {
  const int n    = blockIdx.x;
  const int lane = threadIdx.x & 31;
  const int wv   = threadIdx.x >> 5;
  const int h    = (lane >> 4) & 1;
  const int col  = lane & 15;
  const int dbase = wv * 16;

  v8f c;
  const float bias = Wb[dbase + col];
#pragma unroll
  for (int j = 0; j < 8; ++j) c[j] = bias;

  const float* qrow = q  + (n * TQ + col) * DD;   // A row M = col
  const float* wrow = Ww + (dbase + col) * DD;    // B col d = dbase+col <- Ww row d
#pragma unroll
  for (int ks = 0; ks < DD; ks += 32) {
    v16h a = frag_f32(qrow + ks, h);
    v16h b = frag_f32(wrow + ks, h);
    c = wmma_f16(a, b, c);
  }
  float* o = wq + (n * TQ) * DD + dbase + col;    // C: M=j+8h, N=col
#pragma unroll
  for (int j = 0; j < 8; ++j) o[(j + 8 * h) * DD] = c[j];
}

// --------------- kernel 3: fused uk-GEMM / tanh-score / softmax / x --------
// grid = (KCH, NB); block = 256 threads = 8 waves; wave wv owns d-tile wv of x.
__global__ __launch_bounds__(256) void main_kernel(const float* __restrict__ k,
                                                   const float* __restrict__ mask,
                                                   const float* __restrict__ wq,
                                                   const _Float16* __restrict__ uwh,
                                                   const float* __restrict__ Ub,
                                                   const float* __restrict__ Vw,
                                                   const float* __restrict__ Vb,
                                                   float* __restrict__ part) {
  __shared__ __align__(16) _Float16 kl[IC][KL_STR];   // k chunk, row-major f16
  __shared__ __align__(16) _Float16 kt[DD][KT_STR];   // k chunk, transposed f16
  __shared__ __align__(16) float    wql[TQ][WQ_STR];  // wq[n]
  // Dual-purpose buffer: phase 1 = raw f32 k tile (async DMA target),
  // phase 2 = uk tile (overwrites the raw k after it has been converted).
  __shared__ __align__(16) float    ukl[IC][UK_STR];
  __shared__ __align__(16) _Float16 wA[TQ][WA_STR];   // softmax weights (A frag src)
  __shared__ __align__(16) float    sc[IC][TQ + 1];   // raw scores
  __shared__ __align__(16) float    vwl[DD];
  __shared__ __align__(16) float    mk[IC];

  const int chunk = blockIdx.x;
  const int n     = blockIdx.y;
  const int t     = threadIdx.x;
  const int lane  = t & 31;
  const int wv    = t >> 5;
  const int h     = (lane >> 4) & 1;
  const int col   = lane & 15;

  // Stage wq[n] and Vw once.
  for (int idx = t; idx < TQ * (DD / 4); idx += 256) {
    int r = idx >> 5, c4 = (idx & 31) * 4;
    v4f v = *(const v4f*)(wq + (n * TQ + r) * DD + c4);
    *(v4f*)&wql[r][c4] = v;
  }
  if (t < DD) vwl[t] = Vw[t];
  const float vb = Vb[0];

  // Hoist: Uw B-fragments (invariant over all key tiles) and uk bias.
  const _Float16* urow = uwh + (wv * 16 + col) * DD;
  v16h bU0 = frag_h16(urow,      h);
  v16h bU1 = frag_h16(urow + 32, h);
  v16h bU2 = frag_h16(urow + 64, h);
  v16h bU3 = frag_h16(urow + 96, h);
  const float ubias = Ub[wv * 16 + col];

  v8f xacc;
#pragma unroll
  for (int j = 0; j < 8; ++j) xacc[j] = 0.0f;

  const int qi  = t & 15;        // score-phase mapping
  const int kk0 = t >> 4;        // handles keys kk0 and kk0+16

  for (int ic = 0; ic < CHUNK / IC; ++ic) {
    const int kbase = chunk * CHUNK + ic * IC;
    __syncthreads();  // previous tile fully consumed (ukl readers, kt readers)

    // ---- async DMA the raw f32 k tile straight into LDS (ukl buffer) ------
    for (int idx = t; idx < IC * (DD / 4); idx += 256) {
      int r = idx >> 5, c4 = (idx & 31) * 4;
      const float* g = k + (size_t)(n * TK + kbase + r) * DD + c4;
      async_copy_b128((unsigned)(size_t)(void*)&ukl[r][c4], g);
    }
    if (t < IC) mk[t] = mask[n * TK + kbase + t];
    // Prefetch next tile while async copies are in flight.
    if (ic + 1 < CHUNK / IC && t < IC)
      __builtin_prefetch(k + (size_t)(n * TK + kbase + IC + t) * DD, 0, 0);
    wait_asynccnt0();
    __syncthreads();

    // ---- convert raw f32 tile -> f16 row-major (kl) + transposed (kt) -----
    for (int idx = t; idx < IC * (DD / 4); idx += 256) {
      int r = idx >> 5, c4 = (idx & 31) * 4;
      v4f v = *(const v4f*)&ukl[r][c4];
      v4h hv;
#pragma unroll
      for (int i = 0; i < 4; ++i) hv[i] = (_Float16)v[i];
      *(v4h*)&kl[r][c4] = hv;
#pragma unroll
      for (int i = 0; i < 4; ++i) kt[c4 + i][r] = hv[i];
    }
    __syncthreads();

    // ---- uk = k_tile @ Uw.T + Ub : wave wv does both 16-row halves --------
    // (writes ukl, overwriting the raw k tile that is now dead)
    {
      v8f c0, c1;
#pragma unroll
      for (int j = 0; j < 8; ++j) { c0[j] = ubias; c1[j] = ubias; }
#pragma unroll
      for (int ks = 0; ks < 4; ++ks) {
        v16h b = (ks == 0) ? bU0 : (ks == 1) ? bU1 : (ks == 2) ? bU2 : bU3;
        v16h a0 = frag_h16(&kl[col][ks * 32], h);
        v16h a1 = frag_h16(&kl[16 + col][ks * 32], h);
        c0 = wmma_f16(a0, b, c0);
        c1 = wmma_f16(a1, b, c1);
      }
      __syncthreads();  // all raw-k reads done before overwrite
#pragma unroll
      for (int j = 0; j < 8; ++j) {
        ukl[j + 8 * h][wv * 16 + col]      = c0[j];
        ukl[16 + j + 8 * h][wv * 16 + col] = c1[j];
      }
    }
    __syncthreads();

    // ---- scores: s[q,kk] = Vb + sum_d Vw[d]*tanh(wq[q,d]+uk[kk,d]) --------
    {
      float s0 = 0.0f, s1 = 0.0f;
#pragma unroll 4
      for (int d = 0; d < DD; ++d) {
        float w = wql[qi][d], v = vwl[d];
        s0 += v * fast_tanh(w + ukl[kk0][d]);
        s1 += v * fast_tanh(w + ukl[kk0 + 16][d]);
      }
      s0 += vb + (1.0f - mk[kk0])      * -10000.0f;
      s1 += vb + (1.0f - mk[kk0 + 16]) * -10000.0f;
      sc[kk0][qi] = s0;
      sc[kk0 + 16][qi] = s1;
    }
    __syncthreads();

    // ---- softmax over the TQ axis (local per key column) ------------------
    if (t < IC) {
      float m = -1e30f;
#pragma unroll
      for (int qq = 0; qq < TQ; ++qq) m = fmaxf(m, sc[t][qq]);
      float s = 0.0f;
      float e[TQ];
#pragma unroll
      for (int qq = 0; qq < TQ; ++qq) { e[qq] = __expf(sc[t][qq] - m); s += e[qq]; }
      float r = 1.0f / s;
#pragma unroll
      for (int qq = 0; qq < TQ; ++qq) wA[qq][t] = (_Float16)(e[qq] * r);
    }
    __syncthreads();

    // ---- x[q, dtile] += w(16x32) @ k_tile(32x16) --------------------------
    {
      v16h a = frag_h16(&wA[col][0], h);                 // row q = col
      v16h b = frag_h16(&kt[wv * 16 + col][0], h);       // col d = wv*16+col
      xacc = wmma_f16(a, b, xacc);
    }
  }

  // Partial result for this chunk: part[chunk][n][q][d]
  float* o = part + ((size_t)(chunk * NB + n) * TQ) * DD + wv * 16 + col;
#pragma unroll
  for (int j = 0; j < 8; ++j) o[(j + 8 * h) * DD] = xacc[j];
}

// ----------------------- kernel 4: deterministic reduce --------------------
__global__ __launch_bounds__(256) void reduce_kernel(const float* __restrict__ part,
                                                     float* __restrict__ out) {
  int i = blockIdx.x * 256 + threadIdx.x;
  if (i < NB * TQ * DD) {
    float s = 0.0f;
#pragma unroll
    for (int c = 0; c < KCH; ++c) s += part[(size_t)c * NB * TQ * DD + i];
    out[i] = s;
  }
}

// ---------------------------------------------------------------------------
extern "C" void kernel_launch(void* const* d_in, const int* in_sizes, int n_in,
                              void* d_out, int out_size, void* d_ws, size_t ws_size,
                              hipStream_t stream) {
  (void)in_sizes; (void)n_in; (void)out_size; (void)ws_size;
  const float* q    = (const float*)d_in[0];
  const float* k    = (const float*)d_in[1];
  const float* mask = (const float*)d_in[2];
  const float* Ww   = (const float*)d_in[3];
  const float* Wb   = (const float*)d_in[4];
  const float* Uw   = (const float*)d_in[5];
  const float* Ub   = (const float*)d_in[6];
  const float* Vw   = (const float*)d_in[7];
  const float* Vb   = (const float*)d_in[8];
  float* out = (float*)d_out;

  char* ws = (char*)d_ws;
  float*    wq_ws = (float*)ws;                              // 256 KB
  _Float16* uwh   = (_Float16*)(ws + 262144);                // 32 KB
  float*    part  = (float*)(ws + 262144 + 32768);           // 2 MB

  cvt_uw_kernel<<<(DD * DD + 255) / 256, 256, 0, stream>>>(Uw, uwh);
  wq_kernel<<<NB, 256, 0, stream>>>(q, Ww, Wb, wq_ws);
  main_kernel<<<dim3(KCH, NB), 256, 0, stream>>>(k, mask, wq_ws, uwh, Ub, Vw, Vb, part);
  reduce_kernel<<<(NB * TQ * DD + 255) / 256, 256, 0, stream>>>(part, out);
}
